// Decoder_51548197486708
// MI455X (gfx1250) — matrix operand order repaired and checked by reading, not compiled, not run on hardware
//
#include <hip/hip_runtime.h>
#include <math.h>
#include <stdint.h>

#define B_    256
#define S_    128
#define H_    512
#define E_    512
#define V_    32000
#define X_    1536   // E + 2H (GRU input width)
#define G_    1536   // 3H (gate width)
#define TWOH_ 1024
#define KC_   128    // K-chunk staged into LDS (bf16 elements)
#define KCP_  136    // padded LDS row pitch (272 B -> bank-conflict-free)

typedef float  v2f  __attribute__((ext_vector_type(2)));
typedef float  v8f  __attribute__((ext_vector_type(8)));
typedef __bf16 v8bf __attribute__((ext_vector_type(8)));
typedef __bf16 v16bf __attribute__((ext_vector_type(16)));
typedef unsigned short ushort_t;

// D = A(16x4 f32) * B(4x16 f32) + C, wave32 striped.
__device__ __forceinline__ v8f wmma_f32_k4(v2f a, v2f b, v8f c) {
  return __builtin_amdgcn_wmma_f32_16x16x4_f32(false, a, false, b, (short)0, c,
                                               false, false);
}
// D = A(16x32 bf16) * B(32x16 bf16) + C(f32), wave32 striped.
__device__ __forceinline__ v8f wmma_bf16_k32(v16bf a, v16bf b, v8f c) {
  return __builtin_amdgcn_wmma_f32_16x16x32_bf16(false, a, false, b, (short)0, c,
                                                 false, false);
}

__device__ __forceinline__ unsigned int bf16_rne(float x) {
  unsigned int u = __float_as_uint(x);
  return (u + 0x7FFFu + ((u >> 16) & 1u)) >> 16;  // round-to-nearest-even
}

// A fragment 16x32 bf16: lane holds K-run [k+16h,+16) (2x16B).
__device__ __forceinline__ v16bf load_a_bf16(const ushort_t* p, int k, int half) {
  v8bf lo = *(const v8bf*)(p + k + 16 * half);
  v8bf hi = *(const v8bf*)(p + k + 16 * half + 8);
  v16bf a;
#pragma unroll
  for (int i = 0; i < 8; ++i) { a[i] = lo[i]; a[i + 8] = hi[i]; }
  return a;
}
// B fragment from LDS: 16 contiguous bf16 (two 16B ds loads, 16B-aligned).
__device__ __forceinline__ v16bf load_b_lds(const ushort_t* p) {
  v8bf lo = *(const v8bf*)(p);
  v8bf hi = *(const v8bf*)(p + 8);
  v16bf b;
#pragma unroll
  for (int i = 0; i < 8; ++i) { b[i] = lo[i]; b[i + 8] = hi[i]; }
  return b;
}

// Cooperatively stage a 64-row x KC_-col bf16 strip of W (row pitch ldK) into
// LDS (row pitch KCP_) using CDNA5 async global->LDS copies. 256 threads issue
// 4 x b128 each; wave waits its ASYNCcnt, then workgroup barrier.
__device__ __forceinline__ void stage_strip_async(const ushort_t* __restrict__ row0,
                                                  int ldK, ushort_t* sW) {
  const int tid = threadIdx.x;
#pragma unroll
  for (int i = 0; i < 4; ++i) {
    int e   = (tid + i * 256) * 8;   // element index in 64xKC_ strip
    int r   = e >> 7;                // / KC_
    int col = e & (KC_ - 1);
    unsigned lds_off = (unsigned)(uintptr_t)(&sW[r * KCP_ + col]);
    const ushort_t* g = row0 + (size_t)r * ldK + col;
    asm volatile("global_load_async_to_lds_b128 %0, %1, off"
                 :: "v"(lds_off), "v"(g) : "memory");
  }
  asm volatile("s_wait_asynccnt 0x0" ::: "memory");
  __syncthreads();
}

// ---------------------------------------------------------------- prep ----
__global__ void prep_x_kernel(const int* __restrict__ ids,
                              const float* __restrict__ embed_w,
                              const float* __restrict__ weighted,
                              const int* __restrict__ order,
                              float* __restrict__ x) {
  int t = blockIdx.x * blockDim.x + threadIdx.x;  // B_*X_
  if (t >= B_ * X_) return;
  int b = t / X_, j = t - b * X_;
  float v;
  if (j < E_) v = embed_w[(size_t)ids[b] * E_ + j];
  else        v = (*order != 0) ? weighted[b * TWOH_ + (j - E_)] : 0.0f;
  x[t] = v;
}

__global__ void prep_state_kernel(const float* __restrict__ prev_state,
                                  const float* __restrict__ encoded,
                                  const float* __restrict__ Ws_w,
                                  const float* __restrict__ Ws_b,
                                  const int* __restrict__ order,
                                  float* __restrict__ state) {
  int t = blockIdx.x * blockDim.x + threadIdx.x;  // B_*H_
  if (t >= B_ * H_) return;
  int b = t / H_, h = t - b * H_;
  float v;
  if (*order != 0) {
    v = prev_state[t];
  } else {  // rare path: prev_state = encoded[:, -1] @ Ws_w.T + Ws_b
    const float* e = encoded + ((size_t)b * S_ + (S_ - 1)) * TWOH_;
    const float* w = Ws_w + (size_t)h * TWOH_;
    float acc = Ws_b[h];
    for (int k = 0; k < TWOH_; ++k) acc = fmaf(e[k], w[k], acc);
    v = acc;
  }
  state[t] = v;
}

// --------------------------------------------- f32 -> bf16 quantize pass ----
__global__ void f32_to_bf16_kernel(const float* __restrict__ in,
                                   ushort_t* __restrict__ out, long long n) {
  long long t = ((long long)blockIdx.x * blockDim.x + threadIdx.x) * 8;
  if (t >= n) return;
  float4 f0 = *(const float4*)(in + t);
  float4 f1 = *(const float4*)(in + t + 4);
  uint4 o;
  o.x = bf16_rne(f0.x) | (bf16_rne(f0.y) << 16);
  o.y = bf16_rne(f0.z) | (bf16_rne(f0.w) << 16);
  o.z = bf16_rne(f1.x) | (bf16_rne(f1.y) << 16);
  o.w = bf16_rne(f1.z) | (bf16_rne(f1.w) << 16);
  *(uint4*)(out + t) = o;
}

// ------------------------------------------------- generic WMMA f32 GEMM ----
// C[M,N] = A[M,K] * W[N,K]^T + bias[N].  Wave tile: 32(M) x 64(N). Exact f32.
__global__ void gemm_wmma_f32(const float* __restrict__ A,
                              const float* __restrict__ W,
                              const float* __restrict__ bias,
                              float* __restrict__ C,
                              int M, int N, int K, int tiles_n) {
  const int wave = blockIdx.x * (blockDim.x >> 5) + (threadIdx.x >> 5);
  const int lane = threadIdx.x & 31;
  const int half = lane >> 4;
  const int lidx = lane & 15;
  const int tm = wave / tiles_n;
  const int tn = wave - tm * tiles_n;
  const int m0 = tm * 32;
  const int n0 = tn * 64;
  if (m0 >= M) return;  // wave-uniform

  const float* a0p = A + (size_t)(m0 + lidx) * K + 2 * half;
  const float* a1p = A + (size_t)(m0 + 16 + lidx) * K + 2 * half;
  const float* w0p = W + (size_t)(n0 + lidx) * K + 2 * half;
  const float* w1p = w0p + (size_t)16 * K;
  const float* w2p = w0p + (size_t)32 * K;
  const float* w3p = w0p + (size_t)48 * K;

  v8f c[2][4];
#pragma unroll
  for (int r = 0; r < 2; ++r)
#pragma unroll
    for (int j = 0; j < 4; ++j) c[r][j] = (v8f){0, 0, 0, 0, 0, 0, 0, 0};

  for (int k = 0; k < K; k += 4) {
    if ((k & 63) == 0) {
      __builtin_prefetch(w0p + k + 128, 0, 1);
      __builtin_prefetch(w1p + k + 128, 0, 1);
      __builtin_prefetch(w2p + k + 128, 0, 1);
      __builtin_prefetch(w3p + k + 128, 0, 1);
    }
    v2f a0 = *(const v2f*)(a0p + k);
    v2f a1 = *(const v2f*)(a1p + k);
    v2f b0 = *(const v2f*)(w0p + k);
    v2f b1 = *(const v2f*)(w1p + k);
    v2f b2 = *(const v2f*)(w2p + k);
    v2f b3 = *(const v2f*)(w3p + k);
    c[0][0] = wmma_f32_k4(a0, b0, c[0][0]);
    c[0][1] = wmma_f32_k4(a0, b1, c[0][1]);
    c[0][2] = wmma_f32_k4(a0, b2, c[0][2]);
    c[0][3] = wmma_f32_k4(a0, b3, c[0][3]);
    c[1][0] = wmma_f32_k4(a1, b0, c[1][0]);
    c[1][1] = wmma_f32_k4(a1, b1, c[1][1]);
    c[1][2] = wmma_f32_k4(a1, b2, c[1][2]);
    c[1][3] = wmma_f32_k4(a1, b3, c[1][3]);
  }
#pragma unroll
  for (int j = 0; j < 4; ++j) {
    int col = n0 + 16 * j + lidx;
    float bs = bias ? bias[col] : 0.0f;
#pragma unroll
    for (int i = 0; i < 8; ++i) {
      int r0 = m0 + 8 * half + i;
      C[(size_t)r0 * N + col]        = c[0][j][i] + bs;
      C[(size_t)(r0 + 16) * N + col] = c[1][j][i] + bs;
    }
  }
}

// -------------------- bf16 WMMA GEMM with async-LDS staged weights ---------
// One workgroup per 64-wide N strip, covering ALL of M=256 (8 waves x 32 rows)
// so W is streamed from memory exactly once. Per K-chunk the 64xKC_ bf16 strip
// is staged to LDS with global_load_async_to_lds_b128; B fragments come from
// LDS (bank-conflict-free padded pitch).
__global__ void gemm_wmma_bf16_staged(const ushort_t* __restrict__ A,  // 256 x K
                                      const ushort_t* __restrict__ W,  // N x K
                                      const float* __restrict__ bias,
                                      float* __restrict__ C,
                                      int N, int K) {
  __shared__ __align__(16) ushort_t sW[64 * KCP_];
  const int wave = threadIdx.x >> 5;  // 0..7
  const int lane = threadIdx.x & 31;
  const int half = lane >> 4;
  const int lidx = lane & 15;
  const int m0 = wave * 32;
  const int n0 = blockIdx.x * 64;

  const ushort_t* a0p = A + (size_t)(m0 + lidx) * K;
  const ushort_t* a1p = A + (size_t)(m0 + 16 + lidx) * K;

  v8f c[2][4];
#pragma unroll
  for (int r = 0; r < 2; ++r)
#pragma unroll
    for (int j = 0; j < 4; ++j) c[r][j] = (v8f){0, 0, 0, 0, 0, 0, 0, 0};

  for (int kc = 0; kc < K; kc += KC_) {
    stage_strip_async(W + (size_t)n0 * K + kc, K, sW);
#pragma unroll 2
    for (int k8 = 0; k8 < KC_; k8 += 32) {
      const int k = kc + k8;
      v16bf a0 = load_a_bf16(a0p, k, half);
      v16bf a1 = load_a_bf16(a1p, k, half);
      v16bf b0 = load_b_lds(&sW[(lidx +  0) * KCP_ + k8 + 16 * half]);
      v16bf b1 = load_b_lds(&sW[(lidx + 16) * KCP_ + k8 + 16 * half]);
      v16bf b2 = load_b_lds(&sW[(lidx + 32) * KCP_ + k8 + 16 * half]);
      v16bf b3 = load_b_lds(&sW[(lidx + 48) * KCP_ + k8 + 16 * half]);
      c[0][0] = wmma_bf16_k32(a0, b0, c[0][0]);
      c[0][1] = wmma_bf16_k32(a0, b1, c[0][1]);
      c[0][2] = wmma_bf16_k32(a0, b2, c[0][2]);
      c[0][3] = wmma_bf16_k32(a0, b3, c[0][3]);
      c[1][0] = wmma_bf16_k32(a1, b0, c[1][0]);
      c[1][1] = wmma_bf16_k32(a1, b1, c[1][1]);
      c[1][2] = wmma_bf16_k32(a1, b2, c[1][2]);
      c[1][3] = wmma_bf16_k32(a1, b3, c[1][3]);
    }
    __syncthreads();  // before next chunk overwrites sW
  }
#pragma unroll
  for (int j = 0; j < 4; ++j) {
    int col = n0 + 16 * j + lidx;
    float bs = bias ? bias[col] : 0.0f;
#pragma unroll
    for (int i = 0; i < 8; ++i) {
      int r0 = m0 + 8 * half + i;
      C[(size_t)r0 * N + col]        = c[0][j][i] + bs;
      C[(size_t)(r0 + 16) * N + col] = c[1][j][i] + bs;
    }
  }
}

// ----------------------------------------------------------- GRU gates ----
__global__ void gru_kernel(const float* __restrict__ gi,
                           const float* __restrict__ gh,
                           const float* __restrict__ state,
                           float* __restrict__ hidden) {
  int t = blockIdx.x * blockDim.x + threadIdx.x;  // B_*H_
  if (t >= B_ * H_) return;
  int b = t / H_, h = t - b * H_;
  size_t base = (size_t)b * G_ + h;
  float ir = gi[base], iz = gi[base + H_], in_ = gi[base + 2 * H_];
  float hr = gh[base], hz = gh[base + H_], hn = gh[base + 2 * H_];
  float r = 1.0f / (1.0f + expf(-(ir + hr)));
  float z = 1.0f / (1.0f + expf(-(iz + hz)));
  float n = tanhf(in_ + r * hn);
  hidden[t] = (1.0f - z) * n + z * state[t];
}

// ------- fused: tanh(enc @ Wc^T + b) . hidden -> score_c  (bf16, staged) ----
// One workgroup per batch row b (8 waves x 16 encoder rows = S=128). Wc strips
// are async-staged to LDS once per block (8x less L2 traffic than per-wave
// streaming). The 64 MB tanh intermediate is never materialized.
__global__ void attn_score_bf16_staged(const ushort_t* __restrict__ enc_bf,
                                       const ushort_t* __restrict__ Wc_bf,
                                       const float* __restrict__ Wc_b,
                                       const float* __restrict__ hidden,
                                       const int* __restrict__ src,
                                       float* __restrict__ score_c) {
  __shared__ __align__(16) ushort_t sW[64 * KCP_];
  const int b    = blockIdx.x;
  const int wave = threadIdx.x >> 5;  // 0..7
  const int lane = threadIdx.x & 31;
  const int half = lane >> 4;
  const int lidx = lane & 15;
  const int s0   = wave * 16;

  const ushort_t* ap = enc_bf + ((size_t)b * S_ + s0 + lidx) * TWOH_;

  float part[8];
#pragma unroll
  for (int i = 0; i < 8; ++i) part[i] = 0.0f;

  for (int np = 0; np < 8; ++np) {
    const int nb = np * 64;
    v8f c[4];
#pragma unroll
    for (int j = 0; j < 4; ++j) c[j] = (v8f){0, 0, 0, 0, 0, 0, 0, 0};

    for (int kc = 0; kc < TWOH_; kc += KC_) {
      stage_strip_async(Wc_bf + (size_t)nb * TWOH_ + kc, TWOH_, sW);
#pragma unroll 2
      for (int k8 = 0; k8 < KC_; k8 += 32) {
        v16bf a  = load_a_bf16(ap, kc + k8, half);
        v16bf b0 = load_b_lds(&sW[(lidx +  0) * KCP_ + k8 + 16 * half]);
        v16bf b1 = load_b_lds(&sW[(lidx + 16) * KCP_ + k8 + 16 * half]);
        v16bf b2 = load_b_lds(&sW[(lidx + 32) * KCP_ + k8 + 16 * half]);
        v16bf b3 = load_b_lds(&sW[(lidx + 48) * KCP_ + k8 + 16 * half]);
        c[0] = wmma_bf16_k32(a, b0, c[0]);
        c[1] = wmma_bf16_k32(a, b1, c[1]);
        c[2] = wmma_bf16_k32(a, b2, c[2]);
        c[3] = wmma_bf16_k32(a, b3, c[3]);
      }
      __syncthreads();
    }
#pragma unroll
    for (int j = 0; j < 4; ++j) {
      int n = nb + 16 * j + lidx;  // this lane's output column
      float hb = hidden[(size_t)b * H_ + n];
      float bs = Wc_b[n];
#pragma unroll
      for (int i = 0; i < 8; ++i) part[i] += tanhf(c[j][i] + bs) * hb;
    }
  }
  for (int off = 1; off < 16; off <<= 1)
#pragma unroll
    for (int i = 0; i < 8; ++i) part[i] += __shfl_xor(part[i], off, 32);

  if (lidx == 0) {
#pragma unroll
    for (int i = 0; i < 8; ++i) {
      int s = s0 + 8 * half + i;
      float m = (src[b * S_ + s] == 0) ? -1000.0f : 0.0f;
      score_c[b * S_ + s] = tanhf(part[i] + m);
    }
  }
}

// Exact-f32 fallback of the fused attention-score kernel.
__global__ void attn_score_f32_kernel(const float* __restrict__ enc,
                                      const float* __restrict__ Wc,
                                      const float* __restrict__ Wc_b,
                                      const float* __restrict__ hidden,
                                      const int* __restrict__ src,
                                      float* __restrict__ score_c) {
  const int wave = blockIdx.x * (blockDim.x >> 5) + (threadIdx.x >> 5);
  const int lane = threadIdx.x & 31;
  const int half = lane >> 4;
  const int lidx = lane & 15;
  const int row0 = wave * 16;
  if (row0 >= B_ * S_) return;
  const int b = row0 / S_;
  const int s0 = row0 - b * S_;

  const float* ap = enc + (size_t)(row0 + lidx) * TWOH_ + 2 * half;

  float part[8];
#pragma unroll
  for (int i = 0; i < 8; ++i) part[i] = 0.0f;

  for (int nt = 0; nt < 32; ++nt) {
    const int n = nt * 16 + lidx;
    const float* wp = Wc + (size_t)n * TWOH_ + 2 * half;
    v8f c = (v8f){0, 0, 0, 0, 0, 0, 0, 0};
#pragma unroll 4
    for (int k = 0; k < TWOH_; k += 4) {
      v2f a  = *(const v2f*)(ap + k);
      v2f bb = *(const v2f*)(wp + k);
      c = wmma_f32_k4(a, bb, c);
    }
    float hb = hidden[(size_t)b * H_ + n];
    float bs = Wc_b[n];
#pragma unroll
    for (int i = 0; i < 8; ++i) part[i] += tanhf(c[i] + bs) * hb;
  }
  for (int off = 1; off < 16; off <<= 1)
#pragma unroll
    for (int i = 0; i < 8; ++i) part[i] += __shfl_xor(part[i], off, 32);

  if (lidx == 0) {
#pragma unroll
    for (int i = 0; i < 8; ++i) {
      int s = s0 + 8 * half + i;
      float m = (src[b * S_ + s] == 0) ? -1000.0f : 0.0f;
      score_c[b * S_ + s] = tanhf(part[i] + m);
    }
  }
}

// ---------------------------- softmax + copy mechanism + attention pool ----
__device__ __forceinline__ float blk_reduce_max(float v, float* sm) {
  int tid = threadIdx.x;
  sm[tid] = v; __syncthreads();
  for (int s = 128; s > 0; s >>= 1) {
    if (tid < s) sm[tid] = fmaxf(sm[tid], sm[tid + s]);
    __syncthreads();
  }
  float r = sm[0]; __syncthreads();
  return r;
}
__device__ __forceinline__ float blk_reduce_sum(float v, float* sm) {
  int tid = threadIdx.x;
  sm[tid] = v; __syncthreads();
  for (int s = 128; s > 0; s >>= 1) {
    if (tid < s) sm[tid] += sm[tid + s];
    __syncthreads();
  }
  float r = sm[0]; __syncthreads();
  return r;
}

__global__ void softmax_copy_kernel(float* __restrict__ pred,  // B x V: in score_g, out probs
                                    const float* __restrict__ score_c,
                                    const int* __restrict__ src,
                                    const int* __restrict__ ids,
                                    const float* __restrict__ enc,
                                    float* __restrict__ wout) {  // B x 2H
  __shared__ float sm[256];
  __shared__ float sc[S_];
  const int b = blockIdx.x;
  const int tid = threadIdx.x;
  float* row = pred + (size_t)b * V_;

  float mx = -3.4e38f;
  for (int i = tid; i < V_; i += 256) mx = fmaxf(mx, row[i]);
  if (tid < S_) mx = fmaxf(mx, score_c[b * S_ + tid]);
  mx = blk_reduce_max(mx, sm);

  float sum = 0.0f;
  for (int i = tid; i < V_; i += 256) sum += expf(row[i] - mx);
  if (tid < S_) sum += expf(score_c[b * S_ + tid] - mx);
  sum = blk_reduce_sum(sum, sm);
  const float inv = 1.0f / sum;

  for (int i = tid; i < V_; i += 256) row[i] = expf(row[i] - mx) * inv;
  if (tid < S_) sc[tid] = expf(score_c[b * S_ + tid] - mx) * inv;
  __syncthreads();

  if (tid == 0) {  // deterministic serial scatter of prob_c onto prob_g
    for (int s = 0; s < S_; ++s) row[src[b * S_ + s]] += sc[s];
  }
  const int iid = ids[b];
  float e = (tid < S_ && src[b * S_ + tid] == iid) ? 1.0f : 0.0f;
  float cnt = blk_reduce_sum(e, sm);  // barriers also fence the scatter
  float denom = fmaxf(cnt, 1.0f);
  if (tid < S_) sc[tid] = (src[b * S_ + tid] == iid) ? sc[tid] / denom : 0.0f;
  __syncthreads();

  for (int d = tid; d < TWOH_; d += 256) {
    float acc = 0.0f;
    const float* eb = enc + (size_t)b * S_ * TWOH_ + d;
    for (int s = 0; s < S_; ++s) acc = fmaf(sc[s], eb[(size_t)s * TWOH_], acc);
    wout[(size_t)b * TWOH_ + d] = acc;
  }
}

// ------------------------------------------------------------- launcher ----
extern "C" void kernel_launch(void* const* d_in, const int* in_sizes, int n_in,
                              void* d_out, int out_size, void* d_ws, size_t ws_size,
                              hipStream_t stream) {
  (void)in_sizes; (void)n_in; (void)out_size;

  const int*   ids        = (const int*)  d_in[0];
  const float* encoded    = (const float*)d_in[1];
  const int*   src        = (const int*)  d_in[2];
  const float* prev_state = (const float*)d_in[3];
  const float* weighted   = (const float*)d_in[4];
  const int*   order      = (const int*)  d_in[5];
  const float* embed_w    = (const float*)d_in[6];
  const float* w_ih       = (const float*)d_in[7];
  const float* w_hh       = (const float*)d_in[8];
  const float* b_ih       = (const float*)d_in[9];
  const float* b_hh       = (const float*)d_in[10];
  const float* Ws_w       = (const float*)d_in[11];
  const float* Ws_b       = (const float*)d_in[12];
  const float* Wo_w       = (const float*)d_in[13];
  const float* Wo_b       = (const float*)d_in[14];
  const float* Wc_w       = (const float*)d_in[15];
  const float* Wc_b       = (const float*)d_in[16];

  float* ws       = (float*)d_ws;
  float* x        = ws;                  // B*X
  float* state_in = x + B_ * X_;         // B*H
  float* gi       = state_in + B_ * H_;  // B*G
  float* gh       = gi + B_ * G_;        // B*G
  float* score_c  = gh + B_ * G_;        // B*S
  const size_t f32_elems = (size_t)B_ * X_ + B_ * H_ + 2 * (size_t)B_ * G_ + B_ * S_;

  // bf16 staging (101 MB): enc, Wo, Wc, hidden
  const long long n_enc = (long long)B_ * S_ * TWOH_;  // 33,554,432
  const long long n_wo  = (long long)V_ * H_;          // 16,384,000
  const long long n_wc  = (long long)H_ * TWOH_;       //    524,288
  const long long n_hid = (long long)B_ * H_;          //    131,072
  ushort_t* enc_bf = (ushort_t*)(ws + f32_elems);
  ushort_t* Wo_bf  = enc_bf + n_enc;
  ushort_t* Wc_bf  = Wo_bf + n_wo;
  ushort_t* hid_bf = Wc_bf + n_wc;
  const size_t need = f32_elems * sizeof(float) +
                      (size_t)(n_enc + n_wo + n_wc + n_hid) * sizeof(ushort_t);
  const bool use_bf16 = (ws_size >= need);

  float* pred   = (float*)d_out;             // B x V
  float* hidden = pred + (size_t)B_ * V_;    // B x H
  float* wout   = hidden + (size_t)B_ * H_;  // B x 2H

  prep_x_kernel<<<(B_ * X_) / 256, 256, 0, stream>>>(ids, embed_w, weighted, order, x);
  prep_state_kernel<<<(B_ * H_) / 256, 256, 0, stream>>>(prev_state, encoded, Ws_w,
                                                         Ws_b, order, state_in);

  {  // gi = x @ w_ih^T + b_ih (exact f32 WMMA: recurrent path)
    int tn = G_ / 64, blocks = ((B_ / 32) * tn) / 8;
    gemm_wmma_f32<<<blocks, 256, 0, stream>>>(x, w_ih, b_ih, gi, B_, G_, X_, tn);
  }
  {  // gh = state @ w_hh^T + b_hh
    int tn = G_ / 64, blocks = ((B_ / 32) * tn) / 8;
    gemm_wmma_f32<<<blocks, 256, 0, stream>>>(state_in, w_hh, b_hh, gh, B_, G_, H_, tn);
  }
  gru_kernel<<<(B_ * H_) / 256, 256, 0, stream>>>(gi, gh, state_in, hidden);

  if (use_bf16) {
    // one-shot quantize (pure bandwidth), then wide-K bf16 WMMA GEMMs
    f32_to_bf16_kernel<<<(unsigned)(n_enc / 2048), 256, 0, stream>>>(encoded, enc_bf, n_enc);
    f32_to_bf16_kernel<<<(unsigned)(n_wo  / 2048), 256, 0, stream>>>(Wo_w, Wo_bf, n_wo);
    f32_to_bf16_kernel<<<(unsigned)(n_wc  / 2048), 256, 0, stream>>>(Wc_w, Wc_bf, n_wc);
    f32_to_bf16_kernel<<<(unsigned)(n_hid / 2048), 256, 0, stream>>>(hidden, hid_bf, n_hid);

    // score_g = hidden @ Wo^T + Wo_b (256 x 32000, K=512); Wo streamed once,
    // async-staged to LDS per 64-col strip.
    gemm_wmma_bf16_staged<<<V_ / 64, 256, 0, stream>>>(hid_bf, Wo_bf, Wo_b, pred,
                                                       V_, H_);
    // fused score_c (one block per b; Wc async-staged to LDS per block)
    attn_score_bf16_staged<<<B_, 256, 0, stream>>>(enc_bf, Wc_bf, Wc_b, hidden,
                                                   src, score_c);
  } else {
    {  // exact-f32 fallback
      int tn = V_ / 64, blocks = ((B_ / 32) * tn) / 8;
      gemm_wmma_f32<<<blocks, 256, 0, stream>>>(hidden, Wo_w, Wo_b, pred, B_, V_, H_, tn);
    }
    {
      int blocks = ((B_ * S_) / 16) / 8;
      attn_score_f32_kernel<<<blocks, 256, 0, stream>>>(encoded, Wc_w, Wc_b, hidden,
                                                        src, score_c);
    }
  }

  softmax_copy_kernel<<<B_, 256, 0, stream>>>(pred, score_c, src, ids, encoded, wout);
}
